// MultiScaleMambaBlock_3547642986687
// MI455X (gfx1250) — matrix operand order repaired and checked by reading, not compiled, not run on hardware
//
#include <hip/hip_runtime.h>
#include <hip/hip_bf16.h>
#include <cstdint>

// ---------------------------------------------------------------------------
// MultiScaleMambaBlock for MI455X (gfx1250): bf16 WMMA GEMMs + register scan.
// GEMM tiles are moved with GLOBAL_LOAD_ASYNC_TO_LDS_B128 (ASYNCcnt path,
// bypasses VGPRs) using double-buffered LDS so DMA of tile k+1 overlaps the
// v_wmma_f32_16x16x32_bf16 work on tile k. Scan keeps h[16] per-lane in VGPRs.
// ---------------------------------------------------------------------------

typedef __bf16 bf16_t;
typedef __attribute__((ext_vector_type(16))) __bf16 v16bf;
typedef __attribute__((ext_vector_type(8)))  __bf16 v8bf;
typedef __attribute__((ext_vector_type(8)))  float  v8f;

union Frag { v16bf v; struct { v8bf lo, hi; } h; };

// problem constants (from reference setup_inputs)
#define NS   3
#define BB   8
#define TT   2048
#define DM   512
#define DI   1024
#define DTR  32
#define DST  16
#define DBLW 64          // dtr + 2*ds

// GEMM tiling: block 128x64, 4 waves, wave tile 32x64 (2x4 WMMA tiles)
#define BM   128
#define BN   64
#define BK   32
#define LDAP 40          // padded LDS row stride (elements) -> conflict-free frags

#define EPI_STORE    0
#define EPI_SOFTPLUS 1
#define EPI_ACC      2

// ---------------------------------------------------------------------------
// f32 -> bf16 convert (grid-stride)
__global__ void k_cvt_bf16(const float* __restrict__ in, bf16_t* __restrict__ out, long n) {
    long i = (long)blockIdx.x * blockDim.x + threadIdx.x;
    long st = (long)gridDim.x * blockDim.x;
    for (; i < n; i += st) out[i] = (bf16_t)in[i];
}

// gather decimated rows x[:, ::s, :] -> bf16  (Dm=512, T=2048 fixed)
__global__ void k_gather_bf16(const float* __restrict__ x, bf16_t* __restrict__ out,
                              int Ts, int s, long n) {
    long i = (long)blockIdx.x * blockDim.x + threadIdx.x;
    long st = (long)gridDim.x * blockDim.x;
    for (; i < n; i += st) {
        int c = (int)(i % DM);
        long r = i / DM;
        int t = (int)(r % Ts);
        int b = (int)(r / Ts);
        out[i] = (bf16_t)x[((long)b * TT + (long)t * s) * DM + c];
    }
}

// ---------------------------------------------------------------------------
// C[M,N] = A[M,K] @ W[N,K]^T   (bf16 inputs, f32 accum via WMMA)
// 128 threads = 4 waves; block tile 128x64; wave tile 32x64 (2x4 of 16x16).
// Double-buffered async-to-LDS staging overlaps DMA with WMMA.
__global__ __launch_bounds__(128)
void k_gemm_bf16(const bf16_t* __restrict__ A, int lda,
                 const bf16_t* __restrict__ W, int ldw,
                 float* __restrict__ C, int ldc,
                 int K, int epi, const float* __restrict__ bias,
                 bf16_t* __restrict__ Cbf) {
    __shared__ bf16_t sA[2][BM * LDAP];
    __shared__ bf16_t sB[2][BN * LDAP];

    const int tid  = threadIdx.x;
    const int lane = tid & 31;
    const int wave = tid >> 5;         // 0..3 -> 32 rows each
    const int half = lane >> 4;        // lane half select
    const int l16  = lane & 15;

    const long row0 = (long)blockIdx.x * BM;
    const long col0 = (long)blockIdx.y * BN;

    const uint32_t ldsA0 = (uint32_t)(uintptr_t)(&sA[0][0]);
    const uint32_t ldsB0 = (uint32_t)(uintptr_t)(&sB[0][0]);

    // issue one K-tile's DMA into LDS buffer `buf`; constant trip counts ->
    // no comparisons, no EXEC predication around the async ops.
    auto stage = [&](int k0, int buf) {
        const uint32_t la = ldsA0 + (uint32_t)buf * (BM * LDAP * 2u);
        const uint32_t lb = ldsB0 + (uint32_t)buf * (BN * LDAP * 2u);
        #pragma unroll
        for (int it = 0; it < 4; ++it) {            // A: 128x32 = 512 16B chunks
            const int c2   = (it << 7) + tid;
            const int r    = c2 >> 2;
            const int coff = (c2 & 3) * 8;
            const uint32_t l = la + (uint32_t)(r * LDAP + coff) * 2u;
            const uint64_t ga =
                (uint64_t)(uintptr_t)&A[(row0 + r) * (long)lda + k0 + coff];
            asm volatile("global_load_async_to_lds_b128 %0, %1, off"
                         :: "v"(l), "v"(ga) : "memory");
        }
        #pragma unroll
        for (int it = 0; it < 2; ++it) {            // B: 64x32 = 256 16B chunks
            const int c2   = (it << 7) + tid;
            const int r    = c2 >> 2;
            const int coff = (c2 & 3) * 8;
            const uint32_t l = lb + (uint32_t)(r * LDAP + coff) * 2u;
            const uint64_t ga =
                (uint64_t)(uintptr_t)&W[(col0 + r) * (long)ldw + k0 + coff];
            asm volatile("global_load_async_to_lds_b128 %0, %1, off"
                         :: "v"(l), "v"(ga) : "memory");
        }
    };

    v8f acc[2][4] = {};

    stage(0, 0);
    asm volatile("s_wait_asynccnt 0x0" ::: "memory");
    __syncthreads();

    int cur = 0;
    for (int k0 = 0; k0 < K; k0 += BK) {
        if (k0 + BK < K) stage(k0 + BK, cur ^ 1);   // overlap next tile's DMA
        if (k0 + 2 * BK < K)                        // warm L2 one tile further
            __builtin_prefetch(&A[(row0 + tid) * (long)lda + k0 + 2 * BK], 0, 1);

        const bf16_t* sAr = &sA[cur][0];
        const bf16_t* sBr = &sB[cur][0];
        Frag af[2], bfr[4];
        #pragma unroll
        for (int i = 0; i < 2; ++i) {
            // A 16x32 bf16 frag: lanes 0-15 K={0..7,16..23}, lanes 16-31 K={8..15,24..31}
            const int m = wave * 32 + i * 16 + l16;
            const bf16_t* pa = &sAr[m * LDAP + half * 8];
            af[i].h.lo = *(const v8bf*)pa;
            af[i].h.hi = *(const v8bf*)(pa + 16);
        }
        #pragma unroll
        for (int j = 0; j < 4; ++j) {
            // B 32x16 bf16 frag: lanes 0-15 K={0..7,16..23}, lanes 16-31 K={8..15,24..31}
            const int n = j * 16 + l16;
            const bf16_t* pb = &sBr[n * LDAP + half * 8];
            bfr[j].h.lo = *(const v8bf*)pb;
            bfr[j].h.hi = *(const v8bf*)(pb + 16);
        }
        #pragma unroll
        for (int i = 0; i < 2; ++i)
            #pragma unroll
            for (int j = 0; j < 4; ++j)
                acc[i][j] = __builtin_amdgcn_wmma_f32_16x16x32_bf16(
                    false, af[i].v, false, bfr[j].v, (short)0, acc[i][j], false, false);

        // my DMA for buf^1 done + everyone finished reading buf  -> safe swap
        asm volatile("s_wait_asynccnt 0x0" ::: "memory");
        __syncthreads();
        cur ^= 1;
    }

    // epilogue: C/D layout = lane gives N, VGPR r gives M (+8 for lanes 16-31)
    #pragma unroll
    for (int i = 0; i < 2; ++i)
        #pragma unroll
        for (int j = 0; j < 4; ++j)
            #pragma unroll
            for (int r = 0; r < 8; ++r) {
                const long row = row0 + wave * 32 + i * 16 + half * 8 + r;
                const long col = col0 + j * 16 + l16;
                const long idx = row * (long)ldc + col;
                float v = acc[i][j][r];
                if (epi == EPI_SOFTPLUS) {
                    v += bias[col];
                    v = (v > 20.0f) ? v : log1pf(__expf(v));
                }
                if (epi == EPI_ACC) C[idx] += v;
                else                C[idx]  = v;
                if (Cbf) Cbf[idx] = (bf16_t)v;
            }
}

// ---------------------------------------------------------------------------
// causal depthwise conv1d (k=4, left pad) + bias + SiLU; reads xi from xz[:, :DI]
__global__ __launch_bounds__(256)
void k_conv_silu(const float* __restrict__ xz, const float* __restrict__ cw,
                 const float* __restrict__ cb, bf16_t* __restrict__ xc_bf, int Ts) {
    long i = (long)blockIdx.x * blockDim.x + threadIdx.x;
    const int d = (int)(i % DI);
    const long r = i / DI;                 // b*Ts + t
    const int t = (int)(r % Ts);
    const long b = r / Ts;
    float acc = cb[d];
    #pragma unroll
    for (int k = 0; k < 4; ++k) {
        const int tt = t - 3 + k;
        if (tt >= 0)
            acc += xz[(b * Ts + tt) * (long)(2 * DI) + d] * cw[d * 4 + k];
    }
    acc = acc / (1.0f + __expf(-acc));     // SiLU
    xc_bf[i] = (bf16_t)acc;
}

// extract dt_raw = dbl[:, :32] as bf16
__global__ __launch_bounds__(256)
void k_dtr_extract(const float* __restrict__ dbl, bf16_t* __restrict__ out, long n) {
    long i = (long)blockIdx.x * blockDim.x + threadIdx.x;
    if (i >= n) return;
    const long row = i >> 5;
    const int  c   = (int)(i & 31);
    out[i] = (bf16_t)dbl[row * DBLW + c];
}

// ---------------------------------------------------------------------------
// selective scan: one lane per (b, d). h[16] in registers, serial over Ts.
// writes gated output  g = (y + x*D) * silu(z)  as bf16 for the out-proj GEMM.
__global__ __launch_bounds__(256)
void k_scan(const float* __restrict__ dtv, const float* __restrict__ dbl,
            const bf16_t* __restrict__ xc_bf, const float* __restrict__ xz,
            const float* __restrict__ A_log, const float* __restrict__ D_skip,
            bf16_t* __restrict__ g_bf, int Ts) {
    const int gtid = blockIdx.x * blockDim.x + threadIdx.x;
    const int b = gtid >> 10;              // / DI
    const int d = gtid & (DI - 1);

    float a[DST], h[DST];
    #pragma unroll
    for (int s = 0; s < DST; ++s) { a[s] = -__expf(A_log[d * DST + s]); h[s] = 0.0f; }
    const float Dv = D_skip[d];

    for (int t = 0; t < Ts; ++t) {
        const long r   = (long)b * Ts + t;
        const float dt = dtv[r * DI + d];
        const float xv = (float)xc_bf[r * DI + d];
        const float zv = xz[r * (long)(2 * DI) + DI + d];
        const float dtx = dt * xv;
        const float* bc = &dbl[r * DBLW];
        float y = 0.0f;
        #pragma unroll
        for (int s = 0; s < DST; ++s) {
            h[s] = h[s] * __expf(dt * a[s]) + dtx * bc[DTR + s];
            y += h[s] * bc[DTR + DST + s];
        }
        y += xv * Dv;
        const float gate = zv / (1.0f + __expf(-zv));
        g_bf[r * DI + d] = (bf16_t)(y * gate);
    }
}

// ---------------------------------------------------------------------------
// linear interp along time to T=2048 (align_corners=False) -> bf16
__global__ __launch_bounds__(256)
void k_interp(const float* __restrict__ o, bf16_t* __restrict__ out, int Ts) {
    long i = (long)blockIdx.x * blockDim.x + threadIdx.x;
    if (Ts == TT) { out[i] = (bf16_t)o[i]; return; }
    const int c = (int)(i % DM);
    const long r = i / DM;
    const int t = (int)(r % TT);
    const long b = r / TT;
    float pos = (t + 0.5f) * ((float)Ts / (float)TT) - 0.5f;
    pos = fminf(fmaxf(pos, 0.0f), (float)(Ts - 1));
    const int lo = (int)floorf(pos);
    const int hi = min(lo + 1, Ts - 1);
    const float w = pos - (float)lo;
    const float v = o[(b * Ts + lo) * (long)DM + c] * (1.0f - w)
                  + o[(b * Ts + hi) * (long)DM + c] * w;
    out[i] = (bf16_t)v;
}

// fused: h = facc + fusion_b + x; LayerNorm(h) -> out. One block per row.
__global__ __launch_bounds__(256)
void k_layernorm(const float* __restrict__ facc, const float* __restrict__ x,
                 const float* __restrict__ fb, const float* __restrict__ g,
                 const float* __restrict__ be, float* __restrict__ out) {
    const long base = (long)blockIdx.x * DM;
    const int t = threadIdx.x;
    const float h0 = facc[base + t]       + fb[t]       + x[base + t];
    const float h1 = facc[base + 256 + t] + fb[256 + t] + x[base + 256 + t];
    __shared__ float s1[256], s2[256];
    s1[t] = h0 + h1;
    s2[t] = h0 * h0 + h1 * h1;
    __syncthreads();
    for (int st = 128; st > 0; st >>= 1) {
        if (t < st) { s1[t] += s1[t + st]; s2[t] += s2[t + st]; }
        __syncthreads();
    }
    const float mu  = s1[0] * (1.0f / DM);
    const float var = s2[0] * (1.0f / DM) - mu * mu;
    const float inv = rsqrtf(var + 1e-5f);
    out[base + t]       = g[t]       * (h0 - mu) * inv + be[t];
    out[base + 256 + t] = g[256 + t] * (h1 - mu) * inv + be[256 + t];
}

// ---------------------------------------------------------------------------
extern "C" void kernel_launch(void* const* d_in, const int* in_sizes, int n_in,
                              void* d_out, int out_size, void* d_ws, size_t ws_size,
                              hipStream_t stream) {
    const float* x        = (const float*)d_in[0];
    const float* in_w     = (const float*)d_in[1];
    const float* conv_w   = (const float*)d_in[2];
    const float* conv_b   = (const float*)d_in[3];
    const float* xp_w     = (const float*)d_in[4];
    const float* dt_w     = (const float*)d_in[5];
    const float* dt_b     = (const float*)d_in[6];
    const float* A_log    = (const float*)d_in[7];
    const float* D_skip   = (const float*)d_in[8];
    const float* out_w    = (const float*)d_in[9];
    const float* fusion_w = (const float*)d_in[10];
    const float* fusion_b = (const float*)d_in[11];
    const float* ln_g     = (const float*)d_in[12];
    const float* ln_b     = (const float*)d_in[13];
    float* out = (float*)d_out;

    const long MT = (long)BB * TT;   // 16384 rows at full rate

    // workspace carve-up (256B aligned); needs ~360 MB.
    char* ws = (char*)d_ws;
    size_t off = 0;
    auto alloc = [&](size_t bytes) -> char* {
        char* p = ws + off;
        off = (off + bytes + 255) & ~(size_t)255;
        return p;
    };
    bf16_t* wb_in  = (bf16_t*)alloc((size_t)NS * 2 * DI * DM * 2);
    bf16_t* wb_xp  = (bf16_t*)alloc((size_t)NS * DBLW * DI * 2);
    bf16_t* wb_dt  = (bf16_t*)alloc((size_t)NS * DI * DTR * 2);
    bf16_t* wb_out = (bf16_t*)alloc((size_t)NS * DM * DI * 2);
    bf16_t* wb_fu  = (bf16_t*)alloc((size_t)DM * NS * DM * 2);
    bf16_t* xs_bf  = (bf16_t*)alloc((size_t)MT * DM * 2);
    float*  xz     = (float*) alloc((size_t)MT * 2 * DI * 4);
    bf16_t* xc_bf  = (bf16_t*)alloc((size_t)MT * DI * 2);
    float*  dblf   = (float*) alloc((size_t)MT * DBLW * 4);
    bf16_t* dtr_bf = (bf16_t*)alloc((size_t)MT * DTR * 2);
    float*  dtv    = (float*) alloc((size_t)MT * DI * 4);
    bf16_t* g_bf   = (bf16_t*)alloc((size_t)MT * DI * 2);
    float*  o_f    = (float*) alloc((size_t)MT * DM * 4);
    bf16_t* of_bf  = (bf16_t*)alloc((size_t)MT * DM * 2);
    float*  facc   = (float*) alloc((size_t)MT * DM * 4);

    auto cvt = [&](const float* src, bf16_t* dst, long n) {
        k_cvt_bf16<<<dim3((unsigned)((n + 255) / 256)), 256, 0, stream>>>(src, dst, n);
    };
    cvt(in_w,     wb_in,  (long)NS * 2 * DI * DM);
    cvt(xp_w,     wb_xp,  (long)NS * DBLW * DI);
    cvt(dt_w,     wb_dt,  (long)NS * DI * DTR);
    cvt(out_w,    wb_out, (long)NS * DM * DI);
    cvt(fusion_w, wb_fu,  (long)DM * NS * DM);

    auto gemm = [&](const bf16_t* A, int lda, const bf16_t* W, int ldw,
                    float* C, int ldc, long M, int N, int K,
                    int epi, const float* bias, bf16_t* Cbf) {
        dim3 grid((unsigned)(M / BM), (unsigned)(N / BN));
        k_gemm_bf16<<<grid, 128, 0, stream>>>(A, lda, W, ldw, C, ldc, K, epi, bias, Cbf);
    };

    for (int i = 0; i < NS; ++i) {
        const int s  = 1 << i;
        const int Ts = TT >> i;
        const long M = (long)BB * Ts;

        // 1) decimate x[:, ::s, :] -> bf16 A operand
        k_gather_bf16<<<dim3((unsigned)((M * DM + 255) / 256)), 256, 0, stream>>>(
            x, xs_bf, Ts, s, M * DM);
        // 2) in_proj: [M,512] @ [2048,512]^T -> xz f32
        gemm(xs_bf, DM, wb_in + (long)i * 2 * DI * DM, DM, xz, 2 * DI,
             M, 2 * DI, DM, EPI_STORE, nullptr, nullptr);
        // 3) causal conv + SiLU -> xc bf16
        k_conv_silu<<<dim3((unsigned)(M * DI / 256)), 256, 0, stream>>>(
            xz, conv_w + (long)i * DI * 4, conv_b + (long)i * DI, xc_bf, Ts);
        // 4) x_proj: [M,1024] @ [64,1024]^T -> dbl f32 (dt_raw | B | C)
        gemm(xc_bf, DI, wb_xp + (long)i * DBLW * DI, DI, dblf, DBLW,
             M, DBLW, DI, EPI_STORE, nullptr, nullptr);
        // 5) dt_raw -> bf16
        k_dtr_extract<<<dim3((unsigned)(M * DTR / 256)), 256, 0, stream>>>(
            dblf, dtr_bf, M * DTR);
        // 6) dt_proj + softplus(+dt_b): [M,32] @ [1024,32]^T -> dtv f32
        gemm(dtr_bf, DTR, wb_dt + (long)i * DI * DTR, DTR, dtv, DI,
             M, DI, DTR, EPI_SOFTPLUS, dt_b + (long)i * DI, nullptr);
        // 7) selective scan + gate -> g bf16
        k_scan<<<dim3((unsigned)(BB * DI / 256)), 256, 0, stream>>>(
            dtv, dblf, xc_bf, xz, A_log + (long)i * DI * DST, D_skip + (long)i * DI,
            g_bf, Ts);
        // 8) out_proj: [M,1024] @ [512,1024]^T -> o f32
        gemm(g_bf, DI, wb_out + (long)i * DM * DI, DI, o_f, DM,
             M, DM, DI, EPI_STORE, nullptr, nullptr);
        // 9) time interp to T -> bf16
        k_interp<<<dim3((unsigned)(MT * DM / 256)), 256, 0, stream>>>(o_f, of_bf, Ts);
        // 10) fusion slice GEMM, accumulated: facc (+)= o_i @ Wf[:, i*Dm:(i+1)*Dm]^T
        gemm(of_bf, DM, wb_fu + (long)i * DM, NS * DM, facc, DM,
             MT, DM, DM, (i == 0) ? EPI_STORE : EPI_ACC, nullptr, nullptr);
    }

    // residual + fusion bias + LayerNorm
    k_layernorm<<<dim3((unsigned)MT), 256, 0, stream>>>(
        facc, x, fusion_b, ln_g, ln_b, out);
}
